// BilateralGenerator_19207093748637
// MI455X (gfx1250) — compile-verified, matrix operand order checked by reading, not run on hardware
//
#include <hip/hip_runtime.h>
#include <hip/hip_bf16.h>

typedef __attribute__((ext_vector_type(16))) __bf16 v16bf;
typedef __attribute__((ext_vector_type(8)))  float  v8f;
typedef unsigned __attribute__((ext_vector_type(4))) uint32x4;
typedef int      __attribute__((ext_vector_type(8))) int32x8;

#define NWG     32
#define TSTEPS  512
#define BQ      128
#define HH      512
#define NG      2048

#define USE_TDM_STAGE 1   // 1: tensor_load_to_lds DMA; 0: per-lane async b128

__device__ __forceinline__ unsigned pack_bf16x2(float a, float b) {
    unsigned ua = __float_as_uint(a), ub = __float_as_uint(b);
    unsigned ra = (ua + 0x7FFFu + ((ua >> 16) & 1u)) >> 16;
    unsigned rb = (ub + 0x7FFFu + ((ub >> 16) & 1u)) >> 16;
    return (rb << 16) | (ra & 0xFFFFu);
}

__device__ __forceinline__ float sigmoidf_(float x) {
    return 1.0f / (1.0f + __expf(-x));
}
__device__ __forceinline__ float tanhf_(float x) {
    return __builtin_fmaf(2.0f, sigmoidf_(2.0f * x), -1.0f);
}

// ---------------------------------------------------------------------------
// Chunk source/weight parameters; chunk 0 of every layer reads only
// step-stable data (noise or previous-parity h) so it can be pre-issued
// before the layer's grid barrier.
// ---------------------------------------------------------------------------
__device__ __forceinline__ void chunk_params(
    int l, int c, int t, int cur, int prv,
    const unsigned* noiseT, const unsigned* hT,
    const unsigned* Wx0t, const unsigned* Wxrt, const unsigned* Wht,
    const unsigned*& srcT, int& kt0, int& KtSrc,
    const unsigned*& W, int& Kt, int& ktW) {
    if (l == 0) {
        if (c == 0) { srcT = noiseT + (size_t)t * 8192; kt0 = 0; KtSrc = 4;
                      W = Wx0t; Kt = 4; ktW = 0; }
        else        { srcT = hT + (size_t)(prv * 3 + 0) * 32768; kt0 = (c - 1) * 4; KtSrc = 16;
                      W = Wht; Kt = 16; ktW = (c - 1) * 4; }
    } else {
        if (c < 4)  { srcT = hT + (size_t)(prv * 3 + l) * 32768; kt0 = c * 4; KtSrc = 16;
                      W = Wht + (size_t)l * (HH * NG / 2); Kt = 16; ktW = c * 4; }
        else        { srcT = hT + (size_t)(cur * 3 + (l - 1)) * 32768; kt0 = (c - 4) * 4; KtSrc = 16;
                      W = Wxrt + (size_t)(l - 1) * (HH * NG / 2); Kt = 16; ktW = (c - 4) * 4; }
    }
}

// ---------------------------------------------------------------------------
// Stage one 32KB A-fragment chunk into LDS via the Tensor Data Mover:
// one descriptor (wave 0), 2D tile: 8 rows (mt) x 4KB, row stride KtSrc*1KB.
// ---------------------------------------------------------------------------
__device__ __forceinline__ void stage_issue(unsigned ldsAddr, const unsigned* srcT,
                                            int kt0, int KtSrc, int tid, int wave) {
#if USE_TDM_STAGE
    if (wave == 0) {
        unsigned long long ga = (unsigned long long)(const void*)(srcT + kt0 * 256);
        uint32x4 g0 = { 1u,                                    // count=1 user D#
                        ldsAddr,                               // lds_addr
                        (unsigned)ga,                          // global_addr[31:0]
                        (unsigned)((ga >> 32) & 0x01FFFFFFull) // global_addr[56:32]
                            | 0x80000000u };                   // type=2 ("image")
        unsigned stride0 = (unsigned)(KtSrc * 128);            // 8B units
        int32x8 g1 = { 0x00030000,          // wg_mask=0, data_size=3 (8B), flags=0
                       0,                   // atomic barrier addr = 0
                       0x00004000,          // tensor_dim0 = 0x40000000 (hi16 here)
                       0x02004000,          // tensor_dim1 hi16 | tile_dim0=512
                       8,                   // tile_dim1=8, tile_dim2=0
                       (int)stride0,        // tensor_dim0_stride[31:0]
                       0, 0 };
        asm volatile("tensor_load_to_lds %0, %1" :: "s"(g0), "s"(g1) : "memory");
    }
#else
    for (int i = 0; i < 8; ++i) {
        int p = tid + i * 256;
        int tt = p >> 6, r = p & 63;
        int ln = r >> 1, half = r & 1;
        int mt = tt >> 2, ktc = tt & 3;
        int srcTile = mt * KtSrc + kt0 + ktc;
        unsigned long long ga =
            (unsigned long long)(const void*)(srcT + ((srcTile * 32 + ln) * 8 + half * 4));
        unsigned la = ldsAddr + (unsigned)(((tt * 32 + ln) * 8 + half * 4) * 4);
        asm volatile("global_load_async_to_lds_b128 %0, %1, off"
                     :: "v"(la), "v"(ga) : "memory");
    }
#endif
}

__device__ __forceinline__ void stage_wait(bool morePending, int wave) {
#if USE_TDM_STAGE
    if (wave == 0) {
        if (morePending) __builtin_amdgcn_s_wait_tensorcnt(1);
        else             __builtin_amdgcn_s_wait_tensorcnt(0);
    }
#else
    if (morePending) asm volatile("s_wait_asynccnt 0x8" ::: "memory");
    else             asm volatile("s_wait_asynccnt 0x0" ::: "memory");
#endif
}

// ---------------------------------------------------------------------------
__global__ void tile_weights_k(const float* __restrict__ src, unsigned* __restrict__ dst,
                               int K, int N) {
    int total = (K * N) >> 1;
    int tilesK = K >> 5;
    for (int p = blockIdx.x * blockDim.x + threadIdx.x; p < total;
         p += gridDim.x * blockDim.x) {
        int tt = p >> 8, r = p & 255;
        int ln = r >> 3, pv = r & 7;
        int nt = tt / tilesK, kt = tt - nt * tilesK;
        int k = kt * 32 + ((pv & 4) ? 16 : 0) + ((ln & 16) ? 8 : 0) + (pv & 3) * 2;
        int n = nt * 16 + (ln & 15);
        dst[p] = pack_bf16x2(src[k * N + n], src[(k + 1) * N + n]);
    }
}

__global__ void tile_noise_k(const float* __restrict__ noise, unsigned* __restrict__ dst) {
    int p = blockIdx.x * blockDim.x + threadIdx.x;
    if (p >= TSTEPS * 8192) return;
    int t = p >> 13, q = p & 8191;
    int tt = q >> 8, r = q & 255;
    int ln = r >> 3, pv = r & 7;
    int mt = tt >> 2, kt = tt & 3;
    int k = kt * 32 + ((pv & 4) ? 16 : 0) + ((ln & 16) ? 8 : 0) + (pv & 3) * 2;
    int m = mt * 16 + (ln & 15);
    const float* sp = noise + ((size_t)m * TSTEPS + t) * 128 + k;
    dst[p] = pack_bf16x2(sp[0], sp[1]);
}

__global__ void bc_precompute_k(const float* __restrict__ hc, const float* __restrict__ Wc,
                                const float* __restrict__ b, float* __restrict__ dst) {
    int idx = blockIdx.x * blockDim.x + threadIdx.x;
    if (idx >= 3 * BQ * NG) return;
    int l = idx / (BQ * NG);
    int rem = idx - l * BQ * NG;
    int m = rem >> 11, n = rem & (NG - 1);
    const float* hrow = hc + (l * BQ + m) * HH;
    const float* wcol = Wc + l * HH * NG + n;
    float acc = b[l * NG + n];
    for (int k = 0; k < HH; ++k) acc += hrow[k] * wcol[k * NG];
    int nt = n >> 4, mt = m >> 4, v = (m & 15) & 7;
    int ln = (n & 15) + (((m & 15) >= 8) ? 16 : 0);
    dst[(((l * 128 + nt) * 8 + mt) * 32 + ln) * 8 + v] = acc;
}

__global__ void hfin_copy_k(const float* __restrict__ hfin32, float* __restrict__ dst) {
    int idx = blockIdx.x * blockDim.x + threadIdx.x;
    if (idx < 3 * BQ * HH) dst[idx] = hfin32[idx];
}

__device__ __forceinline__ void grid_barrier(unsigned* ctr, unsigned target) {
    __syncthreads();
    if (threadIdx.x == 0) {
        __threadfence();
        atomicAdd(ctr, 1u);
        while (*(volatile unsigned*)ctr < target) {
            __builtin_amdgcn_s_sleep(2);
        }
        __threadfence();
    }
    __syncthreads();
}

// ---------------------------------------------------------------------------
// Persistent coupled-LSTM kernel: 32 WGs x 256 threads (8 waves).
// WG w owns h columns [w*16, w*16+16). Wave j: gate g=j>>1, m-half mh=j&1,
// computing gate n-tile g*32+w for m-tiles mh*4..mh*4+3 (4 accumulators).
// ---------------------------------------------------------------------------
__global__ __launch_bounds__(256, 1)
void lstm_persistent_k(const unsigned* __restrict__ noiseT,
                       const unsigned* __restrict__ Wx0t,
                       const unsigned* __restrict__ Wxrt,
                       const unsigned* __restrict__ Wht,
                       const unsigned* __restrict__ Woutt,
                       const float* __restrict__ Bct,
                       const float* __restrict__ bout,
                       unsigned* __restrict__ hT,
                       float* __restrict__ hfin32,
                       float* __restrict__ cst,
                       float* __restrict__ outs,
                       unsigned* __restrict__ ctr) {
    __shared__ v16bf ldsA[2][8 * 4 * 32];                 // 2 x 32 KB (pipelined)
    __shared__ __align__(32) float ldsG[4 * 8 * 256];     // 32 KB gate exchange

    const int w = blockIdx.x;
    const int tid = threadIdx.x;
    const int wave = tid >> 5, lane = tid & 31;
    const int g = wave >> 1, mh = wave & 1;
    const int ntG = g * 32 + w;           // this wave's gate n-tile
    const unsigned ldsBase0 = (unsigned)(uintptr_t)(void*)&ldsA[0][0];
    const unsigned ldsBase1 = (unsigned)(uintptr_t)(void*)&ldsA[1][0];
    unsigned barIdx = 0;

    // pre-issue very first chunk (t=0, layer 0, noise)
    {
        const unsigned* srcT; int kt0, KtSrc, Kt, ktW; const unsigned* W;
        chunk_params(0, 0, 0, 0, 1, noiseT, hT, Wx0t, Wxrt, Wht,
                     srcT, kt0, KtSrc, W, Kt, ktW);
        stage_issue(ldsBase0, srcT, kt0, KtSrc, tid, wave);
    }

    for (int t = 0; t < TSTEPS; ++t) {
        const int cur = t & 1, prv = cur ^ 1;

        for (int l = 0; l < 3; ++l) {
            v8f acc[4];
            {
                const v8f* bc = (const v8f*)Bct;
#pragma unroll
                for (int mti = 0; mti < 4; ++mti)
                    acc[mti] = bc[((l * 128 + ntG) * 8 + mh * 4 + mti) * 32 + lane];
            }

            const int nchunks = (l == 0) ? 5 : 8;
            for (int c = 0; c < nchunks; ++c) {
                __syncthreads();                 // frees buf[(c+1)&1] for refill
                bool more = (c + 1 < nchunks);
                if (more) {
                    const unsigned* srcT; int kt0, KtSrc, Kt, ktW; const unsigned* W;
                    chunk_params(l, c + 1, t, cur, prv, noiseT, hT, Wx0t, Wxrt, Wht,
                                 srcT, kt0, KtSrc, W, Kt, ktW);
                    stage_issue(((c + 1) & 1) ? ldsBase1 : ldsBase0,
                                srcT, kt0, KtSrc, tid, wave);
                }
                stage_wait(more, wave);
                __syncthreads();                 // chunk c visible to all waves

                const unsigned* srcT; int kt0, KtSrc, Kt, ktW; const unsigned* W;
                chunk_params(l, c, t, cur, prv, noiseT, hT, Wx0t, Wxrt, Wht,
                             srcT, kt0, KtSrc, W, Kt, ktW);
                const v16bf* Wv = (const v16bf*)W;
                const v16bf* bufA = ldsA[c & 1];
#pragma unroll
                for (int ktc = 0; ktc < 4; ++ktc) {
                    v16bf bfrag = Wv[(ntG * Kt + ktW + ktc) * 32 + lane];
#pragma unroll
                    for (int mti = 0; mti < 4; ++mti) {
                        v16bf afrag = bufA[((mh * 4 + mti) * 4 + ktc) * 32 + lane];
                        acc[mti] = __builtin_amdgcn_wmma_f32_16x16x32_bf16(
                            false, afrag, false, bfrag, (short)0, acc[mti], false, false);
                    }
                }
            }

            // gate exchange through LDS: tile (g, mt) at [(g*8+mt)*32 + lane]
            __syncthreads();
#pragma unroll
            for (int mti = 0; mti < 4; ++mti)
                ((v8f*)ldsG)[(g * 8 + mh * 4 + mti) * 32 + lane] = acc[mti];
            __syncthreads();

            // pre-issue chunk 0 of next layer / next step (stable sources)
            if (l < 2) {
                const unsigned* srcT; int kt0, KtSrc, Kt, ktW; const unsigned* W;
                chunk_params(l + 1, 0, t, cur, prv, noiseT, hT, Wx0t, Wxrt, Wht,
                             srcT, kt0, KtSrc, W, Kt, ktW);
                stage_issue(ldsBase0, srcT, kt0, KtSrc, tid, wave);
            } else if (t + 1 < TSTEPS) {
                const unsigned* srcT; int kt0, KtSrc, Kt, ktW; const unsigned* W;
                chunk_params(0, 0, t + 1, prv, cur, noiseT, hT, Wx0t, Wxrt, Wht,
                             srcT, kt0, KtSrc, W, Kt, ktW);
                stage_issue(ldsBase0, srcT, kt0, KtSrc, tid, wave);
            }

            // elementwise LSTM cell on this WG's 16 exclusive columns
            unsigned* hOutT = hT + (size_t)(cur * 3 + l) * 32768;
            float* cl = cst + (size_t)l * BQ * HH;
            for (int r = 0; r < 4; ++r) {
                int e = r * 256 + tid;          // 0..1023 = 128 rows x 8 col-pairs
                int m = e >> 3;
                int np = e & 7;
                int nl0 = np * 2, nl1 = nl0 + 1;
                int mt = m >> 4, v = (m & 15) & 7;
                int hi = ((m & 15) >= 8) ? 16 : 0;
                int ln0 = nl0 + hi, ln1 = nl1 + hi;

                float i0 = ldsG[(0 * 8 + mt) * 256 + ln0 * 8 + v];
                float f0 = ldsG[(1 * 8 + mt) * 256 + ln0 * 8 + v];
                float g0 = ldsG[(2 * 8 + mt) * 256 + ln0 * 8 + v];
                float o0 = ldsG[(3 * 8 + mt) * 256 + ln0 * 8 + v];
                float i1 = ldsG[(0 * 8 + mt) * 256 + ln1 * 8 + v];
                float f1 = ldsG[(1 * 8 + mt) * 256 + ln1 * 8 + v];
                float g1 = ldsG[(2 * 8 + mt) * 256 + ln1 * 8 + v];
                float o1 = ldsG[(3 * 8 + mt) * 256 + ln1 * 8 + v];

                int k = w * 16 + nl0;           // even global h column
                float2* cp = (float2*)(cl + (size_t)m * HH + k);
                float2 cold = *cp;
                float c0 = sigmoidf_(f0) * cold.x + sigmoidf_(i0) * tanhf_(g0);
                float c1 = sigmoidf_(f1) * cold.y + sigmoidf_(i1) * tanhf_(g1);
                float h0 = sigmoidf_(o0) * tanhf_(c0);
                float h1 = sigmoidf_(o1) * tanhf_(c1);
                *cp = make_float2(c0, c1);

                int kt = k >> 5, kl = k & 31;
                int pv = ((kl & 16) ? 4 : 0) + ((kl & 7) >> 1);
                int ln2 = (m & 15) + ((kl & 8) ? 16 : 0);
                hOutT[((mt * 16 + kt) * 32 + ln2) * 8 + pv] = pack_bf16x2(h0, h1);

                if (t == TSTEPS - 1)
                    *(float2*)(hfin32 + (size_t)l * BQ * HH + (size_t)m * HH + k) =
                        make_float2(h0, h1);
            }

            ++barIdx;
            grid_barrier(ctr, barIdx * NWG);
        }

        // output projection from bf16-tiled h (layer 2, current parity):
        // 64 tiles over 32 WGs x 2 waves
        if (wave < 2) {
            int tl = w * 2 + wave;
            int mtO = tl >> 3, ntO = tl & 7;
            const v16bf* h2v = (const v16bf*)(hT + (size_t)(cur * 3 + 2) * 32768);
            const v16bf* Wo = (const v16bf*)Woutt;
            v8f acc = {};
            for (int kt = 0; kt < 16; ++kt) {
                v16bf afrag = h2v[(mtO * 16 + kt) * 32 + lane];
                v16bf bfrag = Wo[(ntO * 16 + kt) * 32 + lane];
                acc = __builtin_amdgcn_wmma_f32_16x16x32_bf16(
                    false, afrag, false, bfrag, (short)0, acc, false, false);
            }
            int n = ntO * 16 + (lane & 15);
            float bo = bout[n];
            int mrow = mtO * 16 + ((lane & 16) ? 8 : 0);
#pragma unroll
            for (int v = 0; v < 8; ++v) {
                float val = sigmoidf_(acc[v] + bo);
                outs[((size_t)(mrow + v) * TSTEPS + t) * 128 + n] = val;
            }
        }
    }
}

// ---------------------------------------------------------------------------
extern "C" void kernel_launch(void* const* d_in, const int* in_sizes, int n_in,
                              void* d_out, int out_size, void* d_ws, size_t ws_size,
                              hipStream_t stream) {
    const float* noise    = (const float*)d_in[0];
    const float* hcoupled = (const float*)d_in[1];
    const float* Wx0      = (const float*)d_in[2];
    const float* Wxr      = (const float*)d_in[3];
    const float* Wh       = (const float*)d_in[4];
    const float* Wc       = (const float*)d_in[5];
    const float* bb       = (const float*)d_in[6];
    const float* Wout     = (const float*)d_in[7];
    const float* bout     = (const float*)d_in[8];
    float* outp = (float*)d_out;

    char* ws = (char*)d_ws;
    size_t off = 0;
    auto carve = [&](size_t bytes) -> void* {
        void* p = ws + off;
        off = (off + bytes + 255) & ~(size_t)255;
        return p;
    };
    unsigned* ctr    = (unsigned*)carve(256);
    unsigned* Wx0t   = (unsigned*)carve((size_t)128 * 2048 / 2 * 4);
    unsigned* Wxrt   = (unsigned*)carve((size_t)2 * 512 * 2048 / 2 * 4);
    unsigned* Wht    = (unsigned*)carve((size_t)3 * 512 * 2048 / 2 * 4);
    unsigned* Woutt  = (unsigned*)carve((size_t)512 * 128 / 2 * 4);
    float*    Bct    = (float*)carve((size_t)3 * 128 * 2048 * 4);
    unsigned* noiseT = (unsigned*)carve((size_t)TSTEPS * 8192 * 4);
    unsigned* hT     = (unsigned*)carve((size_t)2 * 3 * 32768 * 4);
    float*    hfin32 = (float*)carve((size_t)3 * 128 * 512 * 4);
    float*    cstv   = (float*)carve((size_t)3 * 128 * 512 * 4);

    hipMemsetAsync(ctr, 0, 256, stream);
    hipMemsetAsync(hT, 0, (size_t)2 * 3 * 32768 * 4, stream);
    hipMemsetAsync(cstv, 0, (size_t)3 * 128 * 512 * 4, stream);

    tile_weights_k<<<512, 256, 0, stream>>>(Wx0, Wx0t, 128, 2048);
    for (int l = 0; l < 2; ++l)
        tile_weights_k<<<2048, 256, 0, stream>>>(Wxr + (size_t)l * 512 * 2048,
                                                 Wxrt + (size_t)l * 512 * 2048 / 2, 512, 2048);
    for (int l = 0; l < 3; ++l)
        tile_weights_k<<<2048, 256, 0, stream>>>(Wh + (size_t)l * 512 * 2048,
                                                 Wht + (size_t)l * 512 * 2048 / 2, 512, 2048);
    tile_weights_k<<<128, 256, 0, stream>>>(Wout, Woutt, 512, 128);
    tile_noise_k<<<(TSTEPS * 8192 + 255) / 256, 256, 0, stream>>>(noise, noiseT);
    bc_precompute_k<<<(3 * 128 * 2048 + 255) / 256, 256, 0, stream>>>(hcoupled, Wc, bb, Bct);

    lstm_persistent_k<<<NWG, 256, 0, stream>>>(noiseT, Wx0t, Wxrt, Wht, Woutt,
                                               Bct, bout, hT, hfin32, cstv, outp, ctr);

    hfin_copy_k<<<(3 * 128 * 512 + 255) / 256, 256, 0, stream>>>(hfin32,
                                                                 outp + (size_t)128 * 512 * 128);
}
